// SuppressRebalanceFusionModel_76605036692074
// MI455X (gfx1250) — matrix operand order repaired and checked by reading, not compiled, not run on hardware
//
#include <hip/hip_runtime.h>
#include <math.h>

typedef __bf16 bf16_t;
typedef __attribute__((ext_vector_type(16))) __bf16 v16bf;
typedef __attribute__((ext_vector_type(8)))  __bf16 v8bf;
typedef __attribute__((ext_vector_type(8)))  float v8f;

// ---------------- helpers ----------------
static __device__ __forceinline__ bf16_t f2bf(float f) {
  unsigned u = __float_as_uint(f);
  unsigned r = u + 0x7FFFu + ((u >> 16) & 1u);      // round-to-nearest-even
  unsigned short h = (unsigned short)(r >> 16);
  return __builtin_bit_cast(bf16_t, h);
}
static __device__ __forceinline__ float bf2f(bf16_t b) {
  unsigned short s = __builtin_bit_cast(unsigned short, b);
  return __uint_as_float(((unsigned)s) << 16);
}
static __device__ __forceinline__ float gelu_f(float x) {
  return 0.5f * x * (1.0f + erff(x * 0.70710678118654752f));
}
static __device__ __forceinline__ float rng01(unsigned idx, unsigned seed) {
  unsigned h = idx * 0x9E3779B1u + seed * 0x85EBCA77u + 0x165667B1u;
  h ^= h >> 16; h *= 0x85EBCA6Bu; h ^= h >> 13; h *= 0xC2B2AE35u; h ^= h >> 16;
  return (float)(h >> 8) * (1.0f / 16777216.0f);
}
static __device__ __forceinline__ v16bf cat16(v8bf lo, v8bf hi) {
  return __builtin_shufflevector(lo, hi, 0, 1, 2, 3, 4, 5, 6, 7, 8, 9, 10, 11, 12, 13, 14, 15);
}
// low 32 bits of a __shared__ generic pointer == LDS byte offset
static __device__ __forceinline__ unsigned lds_off(const void* p) {
  return (unsigned)(size_t)p;
}
// CDNA5 async global->LDS copy of 16 bytes per lane (ASYNCcnt-tracked)
static __device__ __forceinline__ void async_copy_b128(unsigned lds_addr, const void* gaddr) {
  asm volatile("global_load_async_to_lds_b128 %0, %1, off" :: "v"(lds_addr), "v"(gaddr) : "memory");
}
static __device__ __forceinline__ void wait_async0() {
  asm volatile("s_wait_asynccnt 0x0" ::: "memory");
}

// conv3x3 pad=1 over 256x256, x:[B][Cin][256][256], w:[Cout][Cin][3][3]
static __device__ __forceinline__ float conv3x3_at(const float* __restrict__ x,
                                                   const float* __restrict__ w,
                                                   int b, int cout, int y, int xx, int Cin) {
  float acc = 0.f;
  if (y >= 1 && y <= 254 && xx >= 1 && xx <= 254) {  // interior fast path, no exec churn
    for (int ci = 0; ci < Cin; ++ci) {
      const float* xp = x + ((size_t)(b * Cin + ci)) * 65536 + (y - 1) * 256 + (xx - 1);
      const float* wp = w + ((cout * Cin + ci) * 9);
      #pragma unroll
      for (int dy = 0; dy < 3; ++dy)
        #pragma unroll
        for (int dx = 0; dx < 3; ++dx)
          acc += xp[dy * 256 + dx] * wp[dy * 3 + dx];
    }
    return acc;
  }
  for (int ci = 0; ci < Cin; ++ci) {
    const float* xp = x + ((size_t)(b * Cin + ci)) * 65536;
    const float* wp = w + ((cout * Cin + ci) * 9);
    #pragma unroll
    for (int dy = 0; dy < 3; ++dy) {
      int yy = y + dy - 1;
      if (yy < 0 || yy > 255) continue;
      #pragma unroll
      for (int dx = 0; dx < 3; ++dx) {
        int xc = xx + dx - 1;
        if (xc < 0 || xc > 255) continue;
        acc += xp[yy * 256 + xc] * wp[dy * 3 + dx];
      }
    }
  }
  return acc;
}

// ---------------- stems / BN stats ----------------
__global__ void kdiff_kernel(const float* __restrict__ w, float* __restrict__ tkd) {
  int c = threadIdx.x;
  if (c < 128) {
    float s = 0.f;
    for (int j = 0; j < 9; ++j) s += w[c * 9 + j];
    tkd[c] = 0.7f * s;  // THETA * kdiff
  }
}

// S layout: [set][2][128] ; set0=d_stem, set1=b_stem, set2=cdc
__global__ __launch_bounds__(256) void d_stats_kernel(const float* __restrict__ x,
                                                      const float* __restrict__ w,
                                                      const float* __restrict__ bias,
                                                      float* __restrict__ S) {
  int t = threadIdx.x, c = blockIdx.y, b = blockIdx.z;
  int pix = blockIdx.x * 256 + t, y = pix >> 8, xx = pix & 255;
  float v = conv3x3_at(x, w, b, c, y, xx, 3) + bias[c];
  __shared__ float s0[256], s1[256];
  s0[t] = v; s1[t] = v * v; __syncthreads();
  for (int s = 128; s > 0; s >>= 1) { if (t < s) { s0[t] += s0[t + s]; s1[t] += s1[t + s]; } __syncthreads(); }
  if (t == 0) { atomicAdd(&S[0 * 128 + c], s0[0]); atomicAdd(&S[1 * 128 + c], s1[0]); }
}

__global__ __launch_bounds__(256) void b_stats_kernel(const float* __restrict__ x,
                                                      const float* __restrict__ wst, const float* __restrict__ bst,
                                                      const float* __restrict__ wcd, const float* __restrict__ bcd,
                                                      const float* __restrict__ tkd, float* __restrict__ S) {
  int t = threadIdx.x, c = blockIdx.y, b = blockIdx.z;
  int pix = blockIdx.x * 256 + t, y = pix >> 8, xx = pix & 255;
  float va = conv3x3_at(x, wst, b, c, y, xx, 1) + bst[c];
  float xc = x[(size_t)b * 65536 + y * 256 + xx];
  float vc = conv3x3_at(x, wcd, b, c, y, xx, 1) + bcd[c] - tkd[c] * xc;
  __shared__ float s0[256], s1[256];
  s0[t] = va; s1[t] = va * va; __syncthreads();
  for (int s = 128; s > 0; s >>= 1) { if (t < s) { s0[t] += s0[t + s]; s1[t] += s1[t + s]; } __syncthreads(); }
  if (t == 0) { atomicAdd(&S[2 * 128 + c], s0[0]); atomicAdd(&S[3 * 128 + c], s1[0]); }
  __syncthreads();
  s0[t] = vc; s1[t] = vc * vc; __syncthreads();
  for (int s = 128; s > 0; s >>= 1) { if (t < s) { s0[t] += s0[t + s]; s1[t] += s1[t + s]; } __syncthreads(); }
  if (t == 0) { atomicAdd(&S[4 * 128 + c], s0[0]); atomicAdd(&S[5 * 128 + c], s1[0]); }
}

__global__ void finalize_kernel(const float* __restrict__ S, float* __restrict__ MI) {
  int i = threadIdx.x;
  if (i < 384) {
    int set = i >> 7, c = i & 127;
    const float inv = 1.0f / 524288.0f;  // B*H*W
    float mean = S[(set * 2) * 128 + c] * inv;
    float var  = S[(set * 2 + 1) * 128 + c] * inv - mean * mean;
    MI[(set * 2) * 128 + c] = mean;
    MI[(set * 2 + 1) * 128 + c] = rsqrtf(var + 1e-5f);
  }
}

// recompute conv, BN+GELU, emit bf16 in im2col layout [m=b*256+tok][k=c*256+py*16+px]
__global__ __launch_bounds__(256) void apply_d_kernel(const float* __restrict__ x,
                                                      const float* __restrict__ w, const float* __restrict__ bias,
                                                      const float* __restrict__ MI,
                                                      const float* __restrict__ gam, const float* __restrict__ bet,
                                                      bf16_t* __restrict__ out) {
  int t = threadIdx.x, c = blockIdx.y, b = blockIdx.z;
  int pix = blockIdx.x * 256 + t, y = pix >> 8, xx = pix & 255;
  float v = conv3x3_at(x, w, b, c, y, xx, 3) + bias[c];
  v = (v - MI[c]) * MI[128 + c] * gam[c] + bet[c];
  v = gelu_f(v);
  int m = b * 256 + (y >> 4) * 16 + (xx >> 4);
  int k = c * 256 + (y & 15) * 16 + (xx & 15);
  out[(size_t)m * 32768 + k] = f2bf(v);
}

__global__ __launch_bounds__(256) void apply_b_kernel(const float* __restrict__ x,
                                                      const float* __restrict__ wst, const float* __restrict__ bst,
                                                      const float* __restrict__ wcd, const float* __restrict__ bcd,
                                                      const float* __restrict__ tkd, const float* __restrict__ MI,
                                                      const float* __restrict__ g1, const float* __restrict__ b1,
                                                      const float* __restrict__ g2, const float* __restrict__ b2,
                                                      bf16_t* __restrict__ out) {
  int t = threadIdx.x, c = blockIdx.y, b = blockIdx.z;
  int pix = blockIdx.x * 256 + t, y = pix >> 8, xx = pix & 255;
  float va = conv3x3_at(x, wst, b, c, y, xx, 1) + bst[c];
  float xc = x[(size_t)b * 65536 + y * 256 + xx];
  float vc = conv3x3_at(x, wcd, b, c, y, xx, 1) + bcd[c] - tkd[c] * xc;
  va = gelu_f((va - MI[256 + c]) * MI[384 + c] * g1[c] + b1[c]);
  vc = gelu_f((vc - MI[512 + c]) * MI[640 + c] * g2[c] + b2[c]);
  int m = b * 256 + (y >> 4) * 16 + (xx >> 4);
  int k = c * 256 + (y & 15) * 16 + (xx & 15);
  out[(size_t)m * 32768 + k] = f2bf(va + vc);
}

// ---------------- weight bf16 convert (keep native [N][K] layout, pad K) ----------------
__global__ void wcvt_kernel(const float* __restrict__ W, bf16_t* __restrict__ out, int N, int K, int Kpad) {
  size_t i = (size_t)blockIdx.x * blockDim.x + threadIdx.x;
  size_t total = (size_t)N * Kpad;
  if (i < total) {
    int k = (int)(i % Kpad), n = (int)(i / Kpad);
    float v = (k < K) ? W[(size_t)n * K + k] : 0.f;
    out[i] = f2bf(v);
  }
}

// ---------------- generic bf16 WMMA GEMM, C = act(A*Bt^T + bias) [*rowScale] [+beta*Extra] ----
// A: [M][K] row-major bf16, Bt: [N][K] row-major bf16 (natural weight layout).
// M % 64 == 0, N % 64 == 0, K % 32 == 0 required (guaranteed by caller).
// Double-buffered LDS; next tile staged with CDNA5 async global->LDS copies overlapped
// with WMMA on the current tile; synced with s_wait_asynccnt + barrier.
#define BM 64
#define BN 64
#define BK 32
__global__ __launch_bounds__(256)
void gemm_bf16_kernel(const bf16_t* __restrict__ A, const bf16_t* __restrict__ Bt,
                      const float* __restrict__ bias, const float* __restrict__ rowScale,
                      const float* __restrict__ addExtra, float addBeta,
                      float* __restrict__ Cf, bf16_t* __restrict__ Cb,
                      int M, int N, int K, int act) {
  __shared__ __align__(16) bf16_t As[2][BM * BK];   // [m][k]
  __shared__ __align__(16) bf16_t Bs[2][BN * BK];   // [n][k]
  int n0 = blockIdx.x * BN, m0 = blockIdx.y * BM;
  int t = threadIdx.x, lane = t & 31, wave = t >> 5;
  int wm = (wave >> 1) * 16;     // 0..48
  int wn = (wave & 1) * 32;      // 0 / 32
  int r = t >> 2, cg = (t & 3) * 8;
  int am = lane & 15, kb = (lane >> 4) * 16;
  int bn = lane & 15, kbb = (lane >> 4) * 16;
  const bf16_t* gA = A  + (size_t)(m0 + r) * K + cg;
  const bf16_t* gB = Bt + (size_t)(n0 + r) * K + cg;
  v8f acc0 = {}; v8f acc1 = {};
  // stage tile 0 asynchronously
  async_copy_b128(lds_off(&As[0][r * BK + cg]), gA);
  async_copy_b128(lds_off(&Bs[0][r * BK + cg]), gB);
  wait_async0();
  __syncthreads();
  int nT = K / BK;
  for (int i = 0; i < nT; ++i) {
    int buf = i & 1;
    if (i + 1 < nT) {  // overlap: async-stage next tile into the other LDS buffer
      const bf16_t* pa = gA + (size_t)(i + 1) * BK;
      const bf16_t* pb = gB + (size_t)(i + 1) * BK;
      async_copy_b128(lds_off(&As[buf ^ 1][r * BK + cg]), pa);
      async_copy_b128(lds_off(&Bs[buf ^ 1][r * BK + cg]), pb);
    }
    if (i + 2 < nT) {  // CDNA5 global_prefetch_b8 two tiles ahead
      __builtin_prefetch(gA + (size_t)(i + 2) * BK, 0, 1);
      __builtin_prefetch(gB + (size_t)(i + 2) * BK, 0, 1);
    }
    // A fragment 16x32: 16 contiguous K values per lane
    const bf16_t* ap = &As[buf][(wm + am) * BK];
    v16bf af = cat16(*(const v8bf*)(ap + kb), *(const v8bf*)(ap + kb + 8));
    // B fragments 32x16: 16 contiguous K values per lane
    const bf16_t* bp0 = &Bs[buf][(wn + bn) * BK + kbb];
    v16bf bf0 = cat16(*(const v8bf*)bp0, *(const v8bf*)(bp0 + 8));
    const bf16_t* bp1 = &Bs[buf][(wn + 16 + bn) * BK + kbb];
    v16bf bf1 = cat16(*(const v8bf*)bp1, *(const v8bf*)(bp1 + 8));
    acc0 = __builtin_amdgcn_wmma_f32_16x16x32_bf16(false, af, false, bf0, (short)0, acc0, false, false);
    acc1 = __builtin_amdgcn_wmma_f32_16x16x32_bf16(false, af, false, bf1, (short)0, acc1, false, false);
    wait_async0();     // next tile fully in LDS
    __syncthreads();
  }
  int mlo = wm + (lane >> 4) * 8, nc = lane & 15;
  #pragma unroll
  for (int rr = 0; rr < 8; ++rr) {
    int gm = m0 + mlo + rr;
    #pragma unroll
    for (int j = 0; j < 2; ++j) {
      int gn = n0 + wn + j * 16 + nc;
      float v = (j == 0) ? acc0[rr] : acc1[rr];
      if (bias) v += bias[gn];
      if (act == 1) v = gelu_f(v);
      if (rowScale) v *= rowScale[gm];
      if (addExtra) v += addBeta * addExtra[(size_t)gm * N + gn];
      if (Cf) Cf[(size_t)gm * N + gn] = v;
      if (Cb) Cb[(size_t)gm * N + gn] = f2bf(v);
    }
  }
}

// ---------------- small fused kernels ----------------
__global__ __launch_bounds__(256) void prior_kernel(const float* __restrict__ mask, float* __restrict__ prior) {
  int m = blockIdx.x, t = threadIdx.x;
  int b = m >> 8, tok = m & 255, ty = tok >> 4, tx = tok & 15, py = t >> 4, px = t & 15;
  float v = mask[(size_t)b * 65536 + (ty * 16 + py) * 256 + (tx * 16 + px)];
  __shared__ float s0[256];
  s0[t] = v; __syncthreads();
  for (int s = 128; s > 0; s >>= 1) { if (t < s) s0[t] += s0[t + s]; __syncthreads(); }
  if (t == 0) prior[m] = s0[0] * (1.f / 256.f);
}

__global__ void dropout_kernel(const float* __restrict__ h, bf16_t* __restrict__ out, int n, unsigned seed) {
  int i = blockIdx.x * blockDim.x + threadIdx.x;
  if (i < n) {
    float keep = (rng01((unsigned)i, seed) < 0.8f) ? 1.25f : 0.0f;
    out[i] = f2bf(h[i] * keep);
  }
}

__global__ void accum_kernel(const float* __restrict__ x, float* __restrict__ s, float* __restrict__ sq,
                             int n, int first) {
  int i = blockIdx.x * blockDim.x + threadIdx.x;
  if (i < n) {
    float v = x[i];
    if (first) { s[i] = v; sq[i] = v * v; }
    else       { s[i] += v; sq[i] += v * v; }
  }
}

__global__ __launch_bounds__(256) void uncert_kernel(const float* __restrict__ s, const float* __restrict__ sq,
                                                     float* __restrict__ unc) {
  int m = blockIdx.x, t = threadIdx.x;
  size_t i = (size_t)m * 256 + t;
  float mean = s[i] * 0.2f;
  float var = sq[i] * 0.2f - mean * mean;
  __shared__ float s0[256];
  s0[t] = var; __syncthreads();
  for (int r = 128; r > 0; r >>= 1) { if (t < r) s0[t] += s0[t + r]; __syncthreads(); }
  if (t == 0) unc[m] = s0[0] * (1.f / 256.f);
}

__global__ void xcat_kernel(const bf16_t* __restrict__ bt, const float* __restrict__ prior,
                            const float* __restrict__ unc, bf16_t* __restrict__ x) {
  int m = blockIdx.x, t = threadIdx.x;  // 288 threads
  bf16_t v;
  if (t < 256) v = bt[(size_t)m * 256 + t];
  else if (t == 256) v = f2bf(prior[m]);
  else if (t == 257) v = f2bf(unc[m]);
  else v = f2bf(0.f);
  x[(size_t)m * 288 + t] = v;
}

__global__ __launch_bounds__(256) void rel3_kernel(const bf16_t* __restrict__ h2, const float* __restrict__ w3,
                                                   const float* __restrict__ b3, const float* __restrict__ prior,
                                                   float* __restrict__ rel, float* __restrict__ logrel) {
  int m = blockIdx.x, t = threadIdx.x;
  float v = bf2f(h2[(size_t)m * 256 + t]) * w3[t];
  __shared__ float s0[256];
  s0[t] = v; __syncthreads();
  for (int s = 128; s > 0; s >>= 1) { if (t < s) s0[t] += s0[t + s]; __syncthreads(); }
  if (t == 0) {
    float sg = 1.f / (1.f + expf(-(s0[0] + b3[0])));
    float r = 0.6f * (1.f - prior[m]) + 0.4f * sg;
    rel[m] = r;
    logrel[m] = logf(fmaxf(r, 1e-6f));
  }
}

// per-head transpose of V: v[m=b*256+tok][h*32+d] -> vT[(b*8+h)*32+d][tok]
__global__ __launch_bounds__(256) void vT_kernel(const bf16_t* __restrict__ v, bf16_t* __restrict__ vT) {
  int m = blockIdx.x, t = threadIdx.x;
  int b = m >> 8, tok = m & 255, h = t >> 5, d0 = t & 31;
  vT[((size_t)((b * 8 + h) * 32 + d0)) * 256 + tok] = v[(size_t)m * 256 + t];
}

// ---------------- attention (single-wave WMMA tiles) ----------------
__global__ __launch_bounds__(32) void attn_scores_kernel(const bf16_t* __restrict__ q, const bf16_t* __restrict__ k,
                                                         const float* __restrict__ logrel, float* __restrict__ scores) {
  int lane = threadIdx.x;
  int k0 = blockIdx.x * 16, q0 = blockIdx.y * 16;
  int z = blockIdx.z, b = z >> 3, h = z & 7;
  int am = lane & 15, kb = (lane >> 4) * 16;
  int bn = lane & 15, kbb = (lane >> 4) * 16;
  const bf16_t* qr = q + ((size_t)(b * 256) + q0 + am) * 256 + h * 32;
  v16bf af = cat16(*(const v8bf*)(qr + kb), *(const v8bf*)(qr + kb + 8));
  const bf16_t* kr = k + ((size_t)(b * 256) + k0 + bn) * 256 + h * 32 + kbb;
  v16bf bfv = cat16(*(const v8bf*)kr, *(const v8bf*)(kr + 8));
  v8f acc = {};
  acc = __builtin_amdgcn_wmma_f32_16x16x32_bf16(false, af, false, bfv, (short)0, acc, false, false);
  int mlo = (lane >> 4) * 8, nc = lane & 15;
  const float scale = 0.17677669529663687f;  // HD^-0.5
  float lr = logrel[b * 256 + k0 + nc];
  #pragma unroll
  for (int r = 0; r < 8; ++r) {
    int qi = q0 + mlo + r;
    scores[((size_t)z * 256 + qi) * 256 + k0 + nc] = acc[r] * scale + lr;
  }
}

__global__ __launch_bounds__(256) void softmax_kernel(const float* __restrict__ scores, bf16_t* __restrict__ attn) {
  int row = blockIdx.x, t = threadIdx.x;
  float v = scores[(size_t)row * 256 + t];
  __shared__ float s0[256];
  s0[t] = v; __syncthreads();
  for (int s = 128; s > 0; s >>= 1) { if (t < s) s0[t] = fmaxf(s0[t], s0[t + s]); __syncthreads(); }
  float mx = s0[0]; __syncthreads();
  float e = expf(v - mx);
  s0[t] = e; __syncthreads();
  for (int s = 128; s > 0; s >>= 1) { if (t < s) s0[t] += s0[t + s]; __syncthreads(); }
  attn[(size_t)row * 256 + t] = f2bf(e / s0[0]);
}

__global__ __launch_bounds__(32) void attn_v_kernel(const bf16_t* __restrict__ attn, const bf16_t* __restrict__ vT,
                                                    bf16_t* __restrict__ out) {
  int lane = threadIdx.x;
  int q0 = blockIdx.x * 16;
  int z = blockIdx.y, b = z >> 3, h = z & 7;
  v8f acc0 = {}, acc1 = {};
  int am = lane & 15, kb = (lane >> 4) * 16;
  int bn = lane & 15, kbb = (lane >> 4) * 16;
  const bf16_t* arow = attn + ((size_t)z * 256 + q0 + am) * 256;
  const bf16_t* vt0 = vT + ((size_t)z * 32 + bn) * 256 + kbb;
  const bf16_t* vt1 = vT + ((size_t)z * 32 + 16 + bn) * 256 + kbb;
  for (int k0 = 0; k0 < 256; k0 += 32) {
    v16bf af = cat16(*(const v8bf*)(arow + k0 + kb), *(const v8bf*)(arow + k0 + kb + 8));
    v16bf b0 = cat16(*(const v8bf*)(vt0 + k0), *(const v8bf*)(vt0 + k0 + 8));
    v16bf b1 = cat16(*(const v8bf*)(vt1 + k0), *(const v8bf*)(vt1 + k0 + 8));
    acc0 = __builtin_amdgcn_wmma_f32_16x16x32_bf16(false, af, false, b0, (short)0, acc0, false, false);
    acc1 = __builtin_amdgcn_wmma_f32_16x16x32_bf16(false, af, false, b1, (short)0, acc1, false, false);
  }
  int mlo = (lane >> 4) * 8, nc = lane & 15;
  #pragma unroll
  for (int r = 0; r < 8; ++r) {
    int qi = q0 + mlo + r;
    size_t base = ((size_t)(b * 256) + qi) * 256 + h * 32;
    out[base + nc] = f2bf(acc0[r]);
    out[base + 16 + nc] = f2bf(acc1[r]);
  }
}

__global__ __launch_bounds__(256) void pool_kernel(const float* __restrict__ fused, float* __restrict__ pooled) {
  int b = blockIdx.x, d = threadIdx.x;
  float s = 0.f;
  for (int n = 0; n < 256; ++n) s += fused[((size_t)(b * 256) + n) * 256 + d];
  pooled[b * 256 + d] = s * (1.f / 256.f);
}

__global__ void cls_kernel(const float* __restrict__ pooled, const float* __restrict__ w,
                           const float* __restrict__ bias, float* __restrict__ out) {
  int t = threadIdx.x;
  if (t < 24) {
    int b = t / 3, c = t % 3;
    float s = bias[c];
    for (int d = 0; d < 256; ++d) s += pooled[b * 256 + d] * w[c * 256 + d];
    out[t] = s;
  }
}

// ---------------- host ----------------
extern "C" void kernel_launch(void* const* d_in, const int* in_sizes, int n_in,
                              void* d_out, int out_size, void* d_ws, size_t ws_size,
                              hipStream_t stream) {
  const float* doppler  = (const float*)d_in[0];
  const float* bmode    = (const float*)d_in[1];
  const float* rmask    = (const float*)d_in[2];
  const float* d_stem_w = (const float*)d_in[3];
  const float* d_stem_b = (const float*)d_in[4];
  const float* d_bn_g   = (const float*)d_in[5];
  const float* d_bn_b   = (const float*)d_in[6];
  const float* d_patch_w= (const float*)d_in[7];
  const float* d_patch_b= (const float*)d_in[8];
  const float* b_stem_w = (const float*)d_in[9];
  const float* b_stem_b = (const float*)d_in[10];
  const float* b_bn_g   = (const float*)d_in[11];
  const float* b_bn_b   = (const float*)d_in[12];
  const float* cdc_w    = (const float*)d_in[13];
  const float* cdc_b    = (const float*)d_in[14];
  const float* cdc_bn_g = (const float*)d_in[15];
  const float* cdc_bn_b = (const float*)d_in[16];
  const float* b_patch_w= (const float*)d_in[17];
  const float* b_patch_b= (const float*)d_in[18];
  const float* uem_w1   = (const float*)d_in[19];
  const float* uem_b1   = (const float*)d_in[20];
  const float* uem_w2   = (const float*)d_in[21];
  const float* uem_b2   = (const float*)d_in[22];
  const float* rel_w1   = (const float*)d_in[23];
  const float* rel_b1   = (const float*)d_in[24];
  const float* rel_w2   = (const float*)d_in[25];
  const float* rel_b2   = (const float*)d_in[26];
  const float* rel_w3   = (const float*)d_in[27];
  const float* rel_b3   = (const float*)d_in[28];
  const float* q_w      = (const float*)d_in[29];
  const float* q_b      = (const float*)d_in[30];
  const float* k_w      = (const float*)d_in[31];
  const float* k_b      = (const float*)d_in[32];
  const float* v_w      = (const float*)d_in[33];
  const float* v_b      = (const float*)d_in[34];
  const float* o_w      = (const float*)d_in[35];
  const float* o_b      = (const float*)d_in[36];
  const float* cls_w    = (const float*)d_in[37];
  const float* cls_b    = (const float*)d_in[38];

  char* base = (char*)d_ws;
  size_t off = 0;
  auto alloc = [&](size_t bytes) -> void* {
    void* p = base + off;
    off = (off + bytes + 255) & ~(size_t)255;
    return p;
  };

  const int M = 2048, D = 256, KP = 32768;
  bf16_t* dA    = (bf16_t*)alloc((size_t)M * KP * 2);
  bf16_t* bA    = (bf16_t*)alloc((size_t)M * KP * 2);
  float*  S     = (float*) alloc(6 * 128 * 4);
  float*  MI    = (float*) alloc(6 * 128 * 4);
  float*  tkd   = (float*) alloc(128 * 4);
  bf16_t* dWp   = (bf16_t*)alloc((size_t)D * KP * 2);
  bf16_t* bWp   = (bf16_t*)alloc((size_t)D * KP * 2);
  bf16_t* u1T   = (bf16_t*)alloc((size_t)D * D * 2);
  bf16_t* u2T   = (bf16_t*)alloc((size_t)D * D * 2);
  bf16_t* r1T   = (bf16_t*)alloc((size_t)D * 288 * 2);
  bf16_t* r2T   = (bf16_t*)alloc((size_t)D * D * 2);
  bf16_t* qWT   = (bf16_t*)alloc((size_t)D * D * 2);
  bf16_t* kWT   = (bf16_t*)alloc((size_t)D * D * 2);
  bf16_t* vWT   = (bf16_t*)alloc((size_t)D * D * 2);
  bf16_t* oWT   = (bf16_t*)alloc((size_t)D * D * 2);
  float*  dtokF = (float*) alloc((size_t)M * D * 4);
  bf16_t* dtokB = (bf16_t*)alloc((size_t)M * D * 2);
  bf16_t* btokB = (bf16_t*)alloc((size_t)M * D * 2);
  float*  hF    = (float*) alloc((size_t)M * D * 4);
  bf16_t* hdB   = (bf16_t*)alloc((size_t)M * D * 2);
  float*  uoutF = (float*) alloc((size_t)M * D * 4);
  float*  usum  = (float*) alloc((size_t)M * D * 4);
  float*  usq   = (float*) alloc((size_t)M * D * 4);
  float*  prior = (float*) alloc(M * 4);
  float*  unc   = (float*) alloc(M * 4);
  float*  rel   = (float*) alloc(M * 4);
  float*  logrel= (float*) alloc(M * 4);
  bf16_t* xcat  = (bf16_t*)alloc((size_t)M * 288 * 2);
  bf16_t* h1B   = (bf16_t*)alloc((size_t)M * D * 2);
  bf16_t* h2B   = (bf16_t*)alloc((size_t)M * D * 2);
  bf16_t* qB    = (bf16_t*)alloc((size_t)M * D * 2);
  bf16_t* kB    = (bf16_t*)alloc((size_t)M * D * 2);
  bf16_t* vB    = (bf16_t*)alloc((size_t)M * D * 2);
  bf16_t* vTB   = (bf16_t*)alloc((size_t)M * D * 2);
  float*  scores= (float*) alloc((size_t)64 * 256 * 256 * 4);
  bf16_t* attnB = (bf16_t*)alloc((size_t)64 * 256 * 256 * 2);
  bf16_t* aoutB = (bf16_t*)alloc((size_t)M * D * 2);
  float*  fused = (float*) alloc((size_t)M * D * 4);
  float*  pooled= (float*) alloc(8 * 256 * 4);
  (void)in_sizes; (void)n_in; (void)out_size; (void)ws_size;

  hipMemsetAsync(S, 0, 6 * 128 * 4, stream);
  kdiff_kernel<<<1, 128, 0, stream>>>(cdc_w, tkd);

  dim3 cgrid(256, 128, 8);
  d_stats_kernel<<<cgrid, 256, 0, stream>>>(doppler, d_stem_w, d_stem_b, S);
  b_stats_kernel<<<cgrid, 256, 0, stream>>>(bmode, b_stem_w, b_stem_b, cdc_w, cdc_b, tkd, S);
  finalize_kernel<<<1, 384, 0, stream>>>(S, MI);
  apply_d_kernel<<<cgrid, 256, 0, stream>>>(doppler, d_stem_w, d_stem_b, MI, d_bn_g, d_bn_b, dA);
  apply_b_kernel<<<cgrid, 256, 0, stream>>>(bmode, b_stem_w, b_stem_b, cdc_w, cdc_b, tkd, MI,
                                            b_bn_g, b_bn_b, cdc_bn_g, cdc_bn_b, bA);

  auto wcvt = [&](const float* W, bf16_t* outp, int N, int K, int Kpad) {
    size_t tot = (size_t)N * Kpad;
    wcvt_kernel<<<(unsigned)((tot + 255) / 256), 256, 0, stream>>>(W, outp, N, K, Kpad);
  };
  wcvt(d_patch_w, dWp, D, KP, KP);
  wcvt(b_patch_w, bWp, D, KP, KP);
  wcvt(uem_w1, u1T, D, D, D);
  wcvt(uem_w2, u2T, D, D, D);
  wcvt(rel_w1, r1T, D, 258, 288);
  wcvt(rel_w2, r2T, D, D, D);
  wcvt(q_w, qWT, D, D, D);
  wcvt(k_w, kWT, D, D, D);
  wcvt(v_w, vWT, D, D, D);
  wcvt(o_w, oWT, D, D, D);

  auto gemm = [&](const bf16_t* A, const bf16_t* Bw, const float* bias, const float* rs,
                  const float* extra, float beta, float* Cf, bf16_t* Cb,
                  int Mm, int Nn, int Kk, int act) {
    dim3 g((unsigned)(Nn / BN), (unsigned)(Mm / BM));
    gemm_bf16_kernel<<<g, 256, 0, stream>>>(A, Bw, bias, rs, extra, beta, Cf, Cb, Mm, Nn, Kk, act);
  };

  // patch "convs" as im2col GEMMs
  gemm(dA, dWp, d_patch_b, nullptr, nullptr, 0.f, dtokF, dtokB, M, D, KP, 0);
  gemm(bA, bWp, b_patch_b, nullptr, nullptr, 0.f, nullptr, btokB, M, D, KP, 0);

  prior_kernel<<<M, 256, 0, stream>>>(rmask, prior);

  // UEM MC passes
  for (int mc = 0; mc < 5; ++mc) {
    gemm(btokB, u1T, uem_b1, nullptr, nullptr, 0.f, hF, nullptr, M, D, D, 1);
    dropout_kernel<<<(M * D + 255) / 256, 256, 0, stream>>>(hF, hdB, M * D, 0x1357u + (unsigned)mc * 2654435761u);
    gemm(hdB, u2T, uem_b2, nullptr, nullptr, 0.f, uoutF, nullptr, M, D, D, 0);
    accum_kernel<<<(M * D + 255) / 256, 256, 0, stream>>>(uoutF, usum, usq, M * D, mc == 0 ? 1 : 0);
  }
  uncert_kernel<<<M, 256, 0, stream>>>(usum, usq, unc);

  // reliability MLP
  xcat_kernel<<<M, 288, 0, stream>>>(btokB, prior, unc, xcat);
  gemm(xcat, r1T, rel_b1, nullptr, nullptr, 0.f, nullptr, h1B, M, D, 288, 1);
  gemm(h1B, r2T, rel_b2, nullptr, nullptr, 0.f, nullptr, h2B, M, D, D, 1);
  rel3_kernel<<<M, 256, 0, stream>>>(h2B, rel_w3, rel_b3, prior, rel, logrel);

  // q/k/v projections (k,v gated by rel)
  gemm(dtokB, qWT, q_b, nullptr, nullptr, 0.f, nullptr, qB, M, D, D, 0);
  gemm(btokB, kWT, k_b, rel, nullptr, 0.f, nullptr, kB, M, D, D, 0);
  gemm(btokB, vWT, v_b, rel, nullptr, 0.f, nullptr, vB, M, D, D, 0);
  vT_kernel<<<M, 256, 0, stream>>>(vB, vTB);

  // attention
  attn_scores_kernel<<<dim3(16, 16, 64), 32, 0, stream>>>(qB, kB, logrel, scores);
  softmax_kernel<<<64 * 256, 256, 0, stream>>>(scores, attnB);
  attn_v_kernel<<<dim3(16, 64), 32, 0, stream>>>(attnB, vTB, aoutB);

  // output projection + bypass, pooling, classifier
  gemm(aoutB, oWT, o_b, nullptr, dtokF, 0.2f, fused, nullptr, M, D, D, 0);
  pool_kernel<<<8, 256, 0, stream>>>(fused, pooled);
  cls_kernel<<<1, 32, 0, stream>>>(pooled, cls_w, cls_b, (float*)d_out);
}
